// HMPLayer_16930761081290
// MI455X (gfx1250) — matrix operand order repaired and checked by reading, not compiled, not run on hardware
//
#include <hip/hip_runtime.h>
#include <stdint.h>

typedef __attribute__((ext_vector_type(16))) _Float16 v16h;
typedef __attribute__((ext_vector_type(8)))  float    v8f;
typedef int hv4i __attribute__((vector_size(16)));   // matches async-LDS builtin param type

#define NN    20000
#define EE    160000
#define CH    64
#define KSEL  2000
#define EVV   32000
#define CSH   576
#define CWAVES 3            // waves per block in the conv kernel (LDS budget)

#ifndef __has_builtin
#define __has_builtin(x) 0
#endif
#if __has_builtin(__builtin_amdgcn_global_load_async_to_lds_b128)
#define HMP_ASYNC_LDS 1
#else
#define HMP_ASYNC_LDS 0
#endif

union FragB { v16h h; uint4 u[2]; };
union FragA { v16h h; unsigned u[8]; };

__device__ __forceinline__ v8f vz8() {
  v8f z = {0.f,0.f,0.f,0.f,0.f,0.f,0.f,0.f}; return z;
}

__device__ __forceinline__ v8f wmma16(v16h a, v16h b, v8f c) {
  return __builtin_amdgcn_wmma_f32_16x16x32_f16(false, a, false, b, (short)0, c, false, false);
}

// A fragment (16x32 f16) from row-major tile (stride in f16 elems, even).
// ISA layout: lane<16 -> M=lane, K in {0..7,16..23}; lane>=16 -> M=lane-16, K in {8..15,24..31}
__device__ __forceinline__ v16h loadA16(const _Float16* T, int stride, int lane) {
  const int m  = lane & 15;
  const int ko = (lane >> 4) << 3;
  const unsigned* p = reinterpret_cast<const unsigned*>(T + m * stride + ko);
  FragA f;
#pragma unroll
  for (int i = 0; i < 4; ++i) { f.u[i] = p[i]; f.u[4 + i] = p[8 + i]; }
  return f.h;
}

// B fragment (32x16 f16) from column-major B storage bT[n*Kdim + k].
__device__ __forceinline__ v16h loadB16(const _Float16* bT, int Kdim, int ncol, int kbase, int lane) {
  const int n  = ncol + (lane & 15);
  const int k0 = kbase + ((lane >> 4) << 3);
  const _Float16* p = bT + (size_t)n * Kdim + k0;
  FragB f;
  f.u[0] = *reinterpret_cast<const uint4*>(p);
  f.u[1] = *reinterpret_cast<const uint4*>(p + 16);
  return f.h;
}

// ---------------------------------------------------------------------------
// weight prep: transpose to f16 column-major, pad fc_w1 K 8->32
// ---------------------------------------------------------------------------
__global__ void hmp_prep_weights(const float* __restrict__ w1, const float* __restrict__ w2,
                                 const float* __restrict__ lw,
                                 _Float16* __restrict__ w1T, _Float16* __restrict__ w2T,
                                 _Float16* __restrict__ linT) {
  int i = blockIdx.x * 256 + threadIdx.x;
  if (i < 64 * 32) { int n = i >> 5, k = i & 31; w1T[i] = (_Float16)((k < 8) ? w1[k * 64 + n] : 0.0f); }
  if (i < CSH * 64) { int n = i / 64, k = i % 64; w2T[i] = (_Float16)w2[(size_t)k * CSH + n]; }
  if (i < 64 * CSH) { int n = i / CSH, k = i % CSH; linT[i] = (_Float16)lw[(size_t)k * 64 + n]; }
}

__global__ void hmp_zero(float* __restrict__ p, size_t n) {
  size_t i = (size_t)blockIdx.x * 256 + threadIdx.x;
  if (i < n) p[i] = 0.0f;
}

// ---------------------------------------------------------------------------
// WMMA edge-convolution: per 32-edge wave-group:
//   ef(32x32 pad) @ w1(32x64) -> relu -> @ w2(64x576)+b2 -> tp(elementwise) -> @ lin(576x64)
//   GEMM2 and GEMM3 fused along the 576 dimension (32-wide chunks via LDS staging).
//   h[dst] rows are gathered with GLOBAL_LOAD_ASYNC_TO_LDS_B128 when available.
// ---------------------------------------------------------------------------
struct WaveLds {
  _Float16 ef[32 * 32];        // GEMM1 A tile (zero padded K)
  _Float16 hid[32 * 72];       // GEMM2 A tile (stride 72)
  float    hdst[32 * 68];      // gathered h[dst] rows, f32 (stride 68)
  float    sh[32 * 12];        // spherical harmonics (9 used)
  _Float16 tps[2][16 * 36];    // tp staging, per subtile (stride 36)
  int      ssrc[32];
  int      sdst[32];
  float    emsk[32];
};

__global__ void hmp_conv_wmma(const float* __restrict__ hnode, const float* __restrict__ pos,
                              const int* __restrict__ srcA, const int* __restrict__ dstA,
                              const float* __restrict__ maskA, int nE,
                              const _Float16* __restrict__ w1T, const float* __restrict__ b1,
                              const _Float16* __restrict__ w2T, const float* __restrict__ b2,
                              const _Float16* __restrict__ linT,
                              float* __restrict__ agg, float* __restrict__ deg) {
  __shared__ WaveLds lds[CWAVES];
  const int wave = threadIdx.x >> 5;
  const int lane = threadIdx.x & 31;
  WaveLds& L = lds[wave];
  const int nGroups = (nE + 31) >> 5;
  const int group = blockIdx.x * CWAVES + wave;
  if (group >= nGroups) return;          // wave-uniform exit, no block barriers used
  const int ebase = group << 5;

  // --- per-edge geometry: one lane per edge ---
  {
    const int e = ebase + lane;
    const bool ok = e < nE;
    int s = ok ? srcA[e] : 0;
    int d = ok ? dstA[e] : 0;
    float em = ok ? (maskA ? maskA[e] : 1.0f) : 0.0f;
    L.ssrc[lane] = s; L.sdst[lane] = d; L.emsk[lane] = em;
    float vx = pos[s * 3 + 0] - pos[d * 3 + 0];
    float vy = pos[s * 3 + 1] - pos[d * 3 + 1];
    float vz = pos[s * 3 + 2] - pos[d * 3 + 2];
    float len = sqrtf(vx * vx + vy * vy + vz * vz + 1e-12f);
    float inv = 1.0f / len;
    float x = vx * inv, y = vy * inv, z = vz * inv;
    const float s3 = 1.7320508075688772f, s15 = 3.872983346207417f, s5 = 2.2360679774997896f;
    float* shr = &L.sh[lane * 12];
    shr[0] = 1.0f;        shr[1] = s3 * x;      shr[2] = s3 * y;
    shr[3] = s3 * z;      shr[4] = s15 * x * y; shr[5] = s15 * y * z;
    shr[6] = 0.5f * s5 * (3.0f * z * z - 1.0f);
    shr[7] = s15 * x * z; shr[8] = 0.5f * s15 * (x * x - y * y);
    // bessel basis with p=6 polynomial envelope
    float rc = fmaxf(len, 1e-9f);
    float u = fminf(fmaxf(len * 0.2f, 0.0f), 1.0f);
    float u2 = u * u, u3 = u2 * u, u6 = u3 * u3;
    float env = 1.0f - 28.0f * u6 + 48.0f * u6 * u - 21.0f * u6 * u2;
    float a0 = sqrtf(0.4f) * env / rc;
    _Float16* efr = &L.ef[lane * 32];
#pragma unroll
    for (int nb = 0; nb < 8; ++nb)
      efr[nb] = (_Float16)(a0 * sinf((float)(nb + 1) * 3.14159265358979f * len * 0.2f));
#pragma unroll
    for (int k2 = 8; k2 < 32; ++k2) efr[k2] = (_Float16)0.0f;
    atomicAdd(&deg[s], em);
  }

  // --- gather h[dst] rows into LDS (async DMA when available) ---
  for (int t = lane; t < 32 * 16; t += 32) {
    int row = t >> 4, q4 = t & 15;
    const float* gp = hnode + (size_t)L.sdst[row] * CH + q4 * 4;
    float* lp = &L.hdst[row * 68 + q4 * 4];
#if HMP_ASYNC_LDS
    __builtin_amdgcn_global_load_async_to_lds_b128(
        (__attribute__((address_space(1))) hv4i*)gp,
        (__attribute__((address_space(3))) hv4i*)lp, 0, 0);
#else
    *reinterpret_cast<float4*>(lp) = *reinterpret_cast<const float4*>(gp);
#endif
  }
#if HMP_ASYNC_LDS
#if __has_builtin(__builtin_amdgcn_s_wait_asynccnt)
  __builtin_amdgcn_s_wait_asynccnt(0);
#else
  asm volatile("s_wait_asynccnt 0x0" ::: "memory");
#endif
#endif

  const int col15 = lane & 15;
  const int rbase = (lane >> 4) ? 8 : 0;

  // --- GEMM1: hid = relu(ef @ w1 + b1) ---
#pragma unroll
  for (int sub = 0; sub < 2; ++sub) {
    v16h a1 = loadA16(&L.ef[sub * 16 * 32], 32, lane);
#pragma unroll
    for (int nt = 0; nt < 4; ++nt) {
      v16h b = loadB16(w1T, 32, nt * 16, 0, lane);
      v8f acc = wmma16(a1, b, vz8());
      float bias = b1[nt * 16 + col15];
#pragma unroll
      for (int r = 0; r < 8; ++r) {
        float hv = fmaxf(acc[r] + bias, 0.0f);
        L.hid[(sub * 16 + rbase + r) * 72 + nt * 16 + col15] = (_Float16)hv;
      }
    }
  }

  // --- preload GEMM2 A fragments ---
  v16h a2[2][2];
#pragma unroll
  for (int sub = 0; sub < 2; ++sub)
#pragma unroll
    for (int ks = 0; ks < 2; ++ks)
      a2[sub][ks] = loadA16(&L.hid[sub * 16 * 72 + ks * 32], 72, lane);

  // --- fused GEMM2 (N) -> tp -> GEMM3 (K) over the 576 dim in 32-chunks ---
  v8f acc3[2][4];
#pragma unroll
  for (int sub = 0; sub < 2; ++sub)
#pragma unroll
    for (int n3 = 0; n3 < 4; ++n3) acc3[sub][n3] = vz8();

  for (int jp = 0; jp < 18; ++jp) {
    if (jp + 1 < 18) __builtin_prefetch(w2T + (size_t)(jp + 1) * 32 * 64, 0, 1);
#pragma unroll
    for (int jj = 0; jj < 2; ++jj) {
      const int jcol = jp * 32 + jj * 16 + col15;
      v16h b20 = loadB16(w2T, 64, jp * 32 + jj * 16, 0, lane);
      v16h b21 = loadB16(w2T, 64, jp * 32 + jj * 16, 32, lane);
      const float bias2 = b2[jcol];
      const int c = jcol / 9, sx = jcol % 9;
#pragma unroll
      for (int sub = 0; sub < 2; ++sub) {
        v8f acc = wmma16(a2[sub][0], b20, vz8());
        acc = wmma16(a2[sub][1], b21, acc);
#pragma unroll
        for (int r = 0; r < 8; ++r) {
          int m = rbase + r;
          int grow = sub * 16 + m;
          float tv = (acc[r] + bias2) * L.hdst[grow * 68 + c] * L.sh[grow * 12 + sx];
          L.tps[sub][m * 36 + jj * 16 + col15] = (_Float16)tv;
        }
      }
    }
#pragma unroll
    for (int n3 = 0; n3 < 4; ++n3) {
      v16h b3 = loadB16(linT, CSH, n3 * 16, jp * 32, lane);
#pragma unroll
      for (int sub = 0; sub < 2; ++sub) {
        v16h a3 = loadA16(L.tps[sub], 36, lane);
        acc3[sub][n3] = wmma16(a3, b3, acc3[sub][n3]);
      }
    }
  }

  // --- masked scatter-add of messages ---
#pragma unroll
  for (int sub = 0; sub < 2; ++sub)
#pragma unroll
    for (int n3 = 0; n3 < 4; ++n3)
#pragma unroll
      for (int r = 0; r < 8; ++r) {
        int m = sub * 16 + rbase + r;
        float val = acc3[sub][n3][r] * L.emsk[m];
        atomicAdd(&agg[(size_t)L.ssrc[m] * CH + n3 * 16 + col15], val);
      }
}

// h_out = agg/max(deg,1) + h_in
__global__ void hmp_finish(const float* __restrict__ agg, const float* __restrict__ deg,
                           const float* __restrict__ hin, float* __restrict__ hout, int nNodes) {
  int i = blockIdx.x * 256 + threadIdx.x;
  if (i >= nNodes * CH) return;
  float d = fmaxf(deg[i >> 6], 1.0f);
  hout[i] = agg[i] / d + hin[i];
}

__global__ void hmp_score(const float* __restrict__ hl, const float* __restrict__ mw1,
                          const float* __restrict__ mb1, const float* __restrict__ mw2,
                          const float* __restrict__ mb2, float* __restrict__ score, int n) {
  __shared__ float W1[16 * 64];
  __shared__ float B1[64];
  __shared__ float W2[64];
  for (int t = threadIdx.x; t < 16 * 64; t += 256) W1[t] = mw1[t];
  for (int t = threadIdx.x; t < 64; t += 256) { B1[t] = mb1[t]; W2[t] = mw2[t]; }
  __syncthreads();
  int i = blockIdx.x * 256 + threadIdx.x;
  if (i >= n) return;
  float hs[16];
#pragma unroll
  for (int t = 0; t < 16; ++t) hs[t] = hl[(size_t)i * CH + t];
  float o = mb2[0];
  for (int j = 0; j < 64; ++j) {
    float a = B1[j];
#pragma unroll
    for (int t = 0; t < 16; ++t) a += hs[t] * W1[t * 64 + j];
    o += fmaxf(a, 0.0f) * W2[j];
  }
  score[i] = 1.0f / (1.0f + expf(-o));
}

// exact top-K via rank counting (score desc, index asc tie-break)
__global__ void hmp_rank(const float* __restrict__ score, int* __restrict__ flag, int n, int Ks) {
  __shared__ float tile[1024];
  int i = blockIdx.x * 256 + threadIdx.x;
  float si = (i < n) ? score[i] : -1e30f;
  int cnt = 0;
  for (int base = 0; base < n; base += 1024) {
    for (int t = threadIdx.x; t < 1024; t += 256)
      tile[t] = (base + t < n) ? score[base + t] : -1e30f;
    __syncthreads();
    int lim = (n - base < 1024) ? (n - base) : 1024;
    for (int t = 0; t < lim; ++t) {
      float sj = tile[t];
      cnt += (sj > si) || (sj == si && (base + t) < i);
    }
    __syncthreads();
  }
  if (i < n) flag[i] = (cnt < Ks) ? 1 : 0;
}

__global__ void hmp_scanA(const int* __restrict__ flag, int* __restrict__ bsum, int n) {
  __shared__ int sh[256];
  int i = blockIdx.x * 256 + threadIdx.x;
  sh[threadIdx.x] = (i < n) ? flag[i] : 0;
  __syncthreads();
  if (threadIdx.x == 0) { int s = 0; for (int t = 0; t < 256; ++t) s += sh[t]; bsum[blockIdx.x] = s; }
}
__global__ void hmp_scanB(int* __restrict__ bsum, int nb) {
  if (threadIdx.x == 0 && blockIdx.x == 0) {
    int run = 0;
    for (int b = 0; b < nb; ++b) { int v = bsum[b]; bsum[b] = run; run += v; }
  }
}
__global__ void hmp_scanC(const int* __restrict__ flag, const int* __restrict__ bsum,
                          const float* __restrict__ score, const float* __restrict__ hl,
                          const float* __restrict__ pos, int* __restrict__ n2m,
                          int* __restrict__ midx, float* __restrict__ hmas,
                          float* __restrict__ pmas, float* __restrict__ m_out, int n, int Ks) {
  __shared__ int off[256];
  int i = blockIdx.x * 256 + threadIdx.x;
  int f = (i < n) ? flag[i] : 0;
  off[threadIdx.x] = f;
  __syncthreads();
  if (threadIdx.x == 0) {
    int run = bsum[blockIdx.x];
    for (int t = 0; t < 256; ++t) { int v = off[t]; off[t] = run; run += v; }
  }
  __syncthreads();
  if (i >= n) return;
  int idx = off[threadIdx.x];
  if (f && idx < Ks) {
    n2m[i] = idx; midx[idx] = i;
    for (int c = 0; c < CH; ++c) hmas[(size_t)idx * CH + c] = hl[(size_t)i * CH + c];
    for (int c = 0; c < 3; ++c) pmas[idx * 3 + c] = pos[i * 3 + c];
    m_out[i] = score[i];
  } else {
    n2m[i] = -1; m_out[i] = 0.0f;
  }
}

// q/k projections, stored f16 padded K=16 -> 32
__global__ void hmp_qk(const float* __restrict__ hmas, const float* __restrict__ wq,
                       const float* __restrict__ wk, _Float16* __restrict__ qh,
                       _Float16* __restrict__ kh, int Kn) {
  int i = blockIdx.x * 256 + threadIdx.x;
  if (i >= Kn * 16) return;
  int k = i >> 4, j = i & 15;
  float aq = 0.0f, ak = 0.0f;
  for (int t = 0; t < 16; ++t) {
    float hv = hmas[(size_t)k * CH + t];
    aq += hv * wq[t * 16 + j];
    ak += hv * wk[t * 16 + j];
  }
  qh[k * 32 + j] = (_Float16)aq; qh[k * 32 + 16 + j] = (_Float16)0.0f;
  kh[k * 32 + j] = (_Float16)ak; kh[k * 32 + 16 + j] = (_Float16)0.0f;
}

__global__ void hmp_adj(const int* __restrict__ src, const int* __restrict__ dst,
                        const int* __restrict__ n2m, float* __restrict__ adj,
                        int* __restrict__ e2s, int* __restrict__ e2d, float* __restrict__ m2,
                        int nE, int Kn) {
  int e = blockIdx.x * 256 + threadIdx.x;
  if (e >= nE) return;
  int sm = n2m[src[e]], dm = n2m[dst[e]];
  bool ok = (sm >= 0) && (dm >= 0);
  e2s[e] = ok ? sm : 0;
  e2d[e] = ok ? dm : 0;
  m2[e] = ok ? 1.0f : 0.0f;
  if (ok) atomicAdd(&adj[(size_t)sm * Kn + dm], 1.0f);
}

// attn = sigmoid(q k^T / 4); A_virtual; attnm written in-place over adj
__global__ void hmp_attn(const _Float16* __restrict__ qh, const _Float16* __restrict__ kh,
                         const float* __restrict__ adj, float* __restrict__ avirt,
                         float* __restrict__ attnm, int Kn) {
  const int wave = threadIdx.x >> 5;
  const int lane = threadIdx.x & 31;
  const int nt = Kn / 16;
  const int tile = blockIdx.x * 4 + wave;
  if (tile >= nt * nt) return;
  const int ti = tile / nt, tj = tile % nt;
  v16h a = loadA16(qh + (size_t)ti * 16 * 32, 32, lane);
  v16h b = loadB16(kh, 32, tj * 16, 0, lane);
  v8f acc = wmma16(a, b, vz8());
  const int col = tj * 16 + (lane & 15);
  const int rb = ti * 16 + ((lane >> 4) ? 8 : 0);
#pragma unroll
  for (int r = 0; r < 8; ++r) {
    int row = rb + r;
    float at = 1.0f / (1.0f + expf(-acc[r] * 0.25f));
    size_t ix = (size_t)row * Kn + col;
    bool exist = (adj[ix] > 0.0f) || (row == col);
    float av = (!exist && at > 0.5f) ? 1.0f : 0.0f;
    avirt[ix] = av;
    attnm[ix] = at * av;
  }
}

__global__ void hmp_hist(const float* __restrict__ v, int* __restrict__ hist, size_t M) {
  __shared__ int hl[1024];
  for (int t = threadIdx.x; t < 1024; t += 256) hl[t] = 0;
  __syncthreads();
  size_t stride = (size_t)gridDim.x * 256;
  for (size_t i = (size_t)blockIdx.x * 256 + threadIdx.x; i < M; i += stride) {
    float x = v[i];
    if (x > 0.0f) {
      int b = (int)((x - 0.5f) * 2048.0f);
      b = (b < 0) ? 0 : ((b > 1023) ? 1023 : b);
      atomicAdd(&hl[b], 1);
    }
  }
  __syncthreads();
  for (int t = threadIdx.x; t < 1024; t += 256)
    if (hl[t]) atomicAdd(&hist[t], hl[t]);
}

__global__ void hmp_thresh(const int* __restrict__ hist, int* __restrict__ thr, int EV) {
  if (threadIdx.x != 0 || blockIdx.x != 0) return;
  int G = 0, bstar = 0;
  for (int b = 1023; b >= 0; --b) {
    int h = hist[b];
    if (G + h >= EV || b == 0) { bstar = b; break; }
    G += h;
  }
  thr[0] = bstar; thr[1] = G;
}

__device__ __forceinline__ void hmp_classify(float x, int bstar, int& g, int& q) {
  if (x > 0.0f) {
    int b = (int)((x - 0.5f) * 2048.0f);
    b = (b < 0) ? 0 : ((b > 1023) ? 1023 : b);
    g += (b > bstar);
    q += (b == bstar);
  }
}

__global__ void hmp_vselA(const float* __restrict__ v, const int* __restrict__ thr,
                          int* __restrict__ part, size_t M) {
  const int bstar = thr[0];
  size_t base = ((size_t)blockIdx.x * 256 + threadIdx.x) * 16;
  int g = 0, q = 0;
  for (int t = 0; t < 16; ++t) { size_t i = base + t; if (i < M) hmp_classify(v[i], bstar, g, q); }
  __shared__ int sg[256], sq[256];
  sg[threadIdx.x] = g; sq[threadIdx.x] = q;
  __syncthreads();
  if (threadIdx.x == 0) {
    int G = 0, Q = 0;
    for (int t = 0; t < 256; ++t) { G += sg[t]; Q += sq[t]; }
    part[blockIdx.x * 2] = G; part[blockIdx.x * 2 + 1] = Q;
  }
}
__global__ void hmp_vselB(int* __restrict__ part, int nb) {
  if (threadIdx.x != 0 || blockIdx.x != 0) return;
  int G = 0, Q = 0;
  for (int b = 0; b < nb; ++b) {
    int g = part[2 * b], q = part[2 * b + 1];
    part[2 * b] = G; part[2 * b + 1] = Q; G += g; Q += q;
  }
}
__global__ void hmp_vselC(const float* __restrict__ v, const int* __restrict__ thr,
                          const int* __restrict__ part, int* __restrict__ e2s,
                          int* __restrict__ e2d, float* __restrict__ m2,
                          size_t M, int Kn, int Eoff, int EV) {
  const int bstar = thr[0], Gtot = thr[1];
  __shared__ int sg[256], sq[256];
  size_t base = ((size_t)blockIdx.x * 256 + threadIdx.x) * 16;
  int g = 0, q = 0;
  for (int t = 0; t < 16; ++t) { size_t i = base + t; if (i < M) hmp_classify(v[i], bstar, g, q); }
  sg[threadIdx.x] = g; sq[threadIdx.x] = q;
  __syncthreads();
  if (threadIdx.x == 0) {
    int G = part[2 * blockIdx.x], Q = part[2 * blockIdx.x + 1];
    for (int t = 0; t < 256; ++t) { int a = sg[t], b = sq[t]; sg[t] = G; sq[t] = Q; G += a; Q += b; }
  }
  __syncthreads();
  int go = sg[threadIdx.x], qo = sq[threadIdx.x];
  for (int t = 0; t < 16; ++t) {
    size_t i = base + t;
    if (i >= M) break;
    float x = v[i];
    if (x <= 0.0f) continue;
    int b = (int)((x - 0.5f) * 2048.0f);
    b = (b < 0) ? 0 : ((b > 1023) ? 1023 : b);
    if (b > bstar) {
      int p = go++;
      e2s[Eoff + p] = (int)(i / Kn); e2d[Eoff + p] = (int)(i % Kn); m2[Eoff + p] = 1.0f;
    } else if (b == bstar) {
      int p = Gtot + qo++;
      if (p < EV) { e2s[Eoff + p] = (int)(i / Kn); e2d[Eoff + p] = (int)(i % Kn); m2[Eoff + p] = 1.0f; }
    }
  }
}

__global__ void hmp_final(const float* __restrict__ hl, const float* __restrict__ hh,
                          const int* __restrict__ n2m, const float* __restrict__ m_out,
                          float* __restrict__ out_h, int n) {
  int i = blockIdx.x * 256 + threadIdx.x;
  if (i >= n * CH) return;
  int node = i >> 6, c = i & 63;
  float mm = m_out[node];
  int mi = n2m[node];
  float hv = (mi >= 0) ? hh[(size_t)mi * CH + c] : 0.0f;
  out_h[i] = (1.0f - mm) * hl[i] + mm * hv;
}

__global__ void hmp_copy(const float* __restrict__ src, float* __restrict__ dst, size_t n) {
  size_t i = (size_t)blockIdx.x * 256 + threadIdx.x;
  if (i < n) dst[i] = src[i];
}

extern "C" void kernel_launch(void* const* d_in, const int* in_sizes, int n_in,
                              void* d_out, int out_size, void* d_ws, size_t ws_size,
                              hipStream_t stream) {
  (void)in_sizes; (void)n_in; (void)out_size; (void)ws_size;
  const float* h    = (const float*)d_in[0];
  const float* pos  = (const float*)d_in[1];
  const int*   eidx = (const int*)d_in[2];
  const float* fcw1 = (const float*)d_in[3];
  const float* fcb1 = (const float*)d_in[4];
  const float* fcw2 = (const float*)d_in[5];
  const float* fcb2 = (const float*)d_in[6];
  const float* linw = (const float*)d_in[7];
  const float* msw1 = (const float*)d_in[8];
  const float* msb1 = (const float*)d_in[9];
  const float* msw2 = (const float*)d_in[10];
  const float* msb2 = (const float*)d_in[11];
  const float* vgwq = (const float*)d_in[12];
  const float* vgwk = (const float*)d_in[13];
  const int* srcA = eidx;
  const int* dstA = eidx + EE;

  float* out_h = (float*)d_out;
  float* out_p = out_h + (size_t)NN * CH;
  float* out_A = out_p + (size_t)NN * 3;
  float* out_m = out_A + (size_t)KSEL * KSEL;

  uintptr_t arena = (uintptr_t)d_ws;
  auto alloc = [&](size_t bytes) -> void* {
    uintptr_t p = (arena + 255) & ~(uintptr_t)255;
    arena = p + bytes;
    return (void*)p;
  };
  _Float16* w1T  = (_Float16*)alloc((size_t)64 * 32 * 2);
  _Float16* w2T  = (_Float16*)alloc((size_t)CSH * 64 * 2);
  _Float16* linT = (_Float16*)alloc((size_t)64 * CSH * 2);
  float* agg1 = (float*)alloc((size_t)NN * CH * 4);
  float* deg1 = (float*)alloc((size_t)NN * 4);
  float* hloc = (float*)alloc((size_t)NN * CH * 4);
  float* scor = (float*)alloc((size_t)NN * 4);
  int*   flag = (int*)alloc((size_t)NN * 4);
  int*   bsum = (int*)alloc(1024 * 4);
  int*   n2m  = (int*)alloc((size_t)NN * 4);
  int*   midx = (int*)alloc((size_t)KSEL * 4);
  float* hmas = (float*)alloc((size_t)KSEL * CH * 4);
  float* pmas = (float*)alloc((size_t)KSEL * 3 * 4);
  _Float16* qh = (_Float16*)alloc((size_t)KSEL * 32 * 2);
  _Float16* kh = (_Float16*)alloc((size_t)KSEL * 32 * 2);
  float* adj  = (float*)alloc((size_t)KSEL * KSEL * 4);   // reused as attn*A_virtual
  int*   hist = (int*)alloc(1024 * 4);
  int*   thr  = (int*)alloc(64);
  int*   e2s  = (int*)alloc((size_t)(EE + EVV) * 4);
  int*   e2d  = (int*)alloc((size_t)(EE + EVV) * 4);
  float* m2   = (float*)alloc((size_t)(EE + EVV) * 4);
  int*   part = (int*)alloc((size_t)2048 * 4);
  float* agg2 = (float*)alloc((size_t)KSEL * CH * 4);
  float* deg2 = (float*)alloc((size_t)KSEL * 4);
  float* hhier = (float*)alloc((size_t)KSEL * CH * 4);

  auto zl = [&](void* p, size_t n) {
    hmp_zero<<<(unsigned)((n + 255) / 256), 256, 0, stream>>>((float*)p, n);
  };

  // stage 0: weights
  hmp_prep_weights<<<144, 256, 0, stream>>>(fcw1, fcw2, linw, w1T, w2T, linT);

  // stage 1: conv on full graph (WMMA)
  zl(agg1, (size_t)NN * CH); zl(deg1, NN);
  {
    const int nG = (EE + 31) / 32;
    hmp_conv_wmma<<<(nG + CWAVES - 1) / CWAVES, 32 * CWAVES, 0, stream>>>(
        h, pos, srcA, dstA, nullptr, EE, w1T, fcb1, w2T, fcb2, linT, agg1, deg1);
  }
  hmp_finish<<<(NN * CH + 255) / 256, 256, 0, stream>>>(agg1, deg1, h, hloc, NN);

  // stage 2: scoring + exact top-K + compaction
  hmp_score<<<(NN + 255) / 256, 256, 0, stream>>>(hloc, msw1, msb1, msw2, msb2, scor, NN);
  hmp_rank<<<(NN + 255) / 256, 256, 0, stream>>>(scor, flag, NN, KSEL);
  hmp_scanA<<<(NN + 255) / 256, 256, 0, stream>>>(flag, bsum, NN);
  hmp_scanB<<<1, 32, 0, stream>>>(bsum, (NN + 255) / 256);
  hmp_scanC<<<(NN + 255) / 256, 256, 0, stream>>>(flag, bsum, scor, hloc, pos,
                                                  n2m, midx, hmas, pmas, out_m, NN, KSEL);

  // stage 3: adjacency, attention (WMMA), virtual edge selection
  hmp_qk<<<(KSEL * 16 + 255) / 256, 256, 0, stream>>>(hmas, vgwq, vgwk, qh, kh, KSEL);
  zl(adj, (size_t)KSEL * KSEL);
  zl(e2s, EE + EVV); zl(e2d, EE + EVV); zl(m2, EE + EVV);
  hmp_adj<<<(EE + 255) / 256, 256, 0, stream>>>(srcA, dstA, n2m, adj, e2s, e2d, m2, EE, KSEL);
  hmp_attn<<<((KSEL / 16) * (KSEL / 16) + 3) / 4, 128, 0, stream>>>(qh, kh, adj, out_A, adj, KSEL);
  zl(hist, 1024); zl(thr, 2);
  hmp_hist<<<1024, 256, 0, stream>>>(adj, hist, (size_t)KSEL * KSEL);
  hmp_thresh<<<1, 32, 0, stream>>>(hist, thr, EVV);
  {
    const size_t M = (size_t)KSEL * KSEL;
    const int nvb = (int)((M + 4095) / 4096);
    hmp_vselA<<<nvb, 256, 0, stream>>>(adj, thr, part, M);
    hmp_vselB<<<1, 32, 0, stream>>>(part, nvb);
    hmp_vselC<<<nvb, 256, 0, stream>>>(adj, thr, part, e2s, e2d, m2, M, KSEL, EE, EVV);
  }

  // stage 4: conv on master graph (real + virtual edges, WMMA)
  zl(agg2, (size_t)KSEL * CH); zl(deg2, KSEL);
  {
    const int nE2 = EE + EVV;
    const int nG2 = (nE2 + 31) / 32;
    hmp_conv_wmma<<<(nG2 + CWAVES - 1) / CWAVES, 32 * CWAVES, 0, stream>>>(
        hmas, pmas, e2s, e2d, m2, nE2, w1T, fcb1, w2T, fcb2, linT, agg2, deg2);
  }
  hmp_finish<<<(KSEL * CH + 255) / 256, 256, 0, stream>>>(agg2, deg2, hmas, hhier, KSEL);

  // stage 5: combine + outputs
  hmp_final<<<(NN * CH + 255) / 256, 256, 0, stream>>>(hloc, hhier, n2m, out_m, out_h, NN);
  hmp_copy<<<(NN * 3 + 255) / 256, 256, 0, stream>>>(pos, out_p, (size_t)NN * 3);
}
